// RelationEmbedding_42339787604117
// MI455X (gfx1250) — matrix operand order repaired and checked by reading, not compiled, not run on hardware
//
#include <hip/hip_runtime.h>
#include <hip/hip_bf16.h>

// ---------------------------------------------------------------------------
// Types matching the probe-confirmed WMMA builtin signature:
//   __builtin_amdgcn_wmma_f32_16x16x32_bf16(bool, v16bf, bool, v16bf,
//                                           short, v8f, bool, bool)
// ---------------------------------------------------------------------------
typedef __attribute__((ext_vector_type(16))) __bf16 v16bf;
typedef __attribute__((ext_vector_type(8)))  float  v8f;
typedef __attribute__((ext_vector_type(4)))  int    v4i;

#define B_SZ      1024
#define D_HALF    2048
#define D_IN      4096
#define D_OUT     1024
#define N_LABELS  128

#define TILE_K    32
#define LDS_LDK   40      // padded row stride (elements): conflict-free ds_load_b128

// ---------------------------------------------------------------------------
// Async global->LDS (CDNA5 GLOBAL_LOAD_ASYNC_TO_LDS_B128, ASYNCcnt-tracked).
// Builtin signature (from hipcc diagnostic): (v4i AS1*, v4i AS3*, imm, imm).
// Guarded: falls back to VGPR round-trip copy if the builtin is absent.
// ---------------------------------------------------------------------------
#if __has_builtin(__builtin_amdgcn_global_load_async_to_lds_b128)
#define REL_HAS_ASYNC_LDS 1
#else
#define REL_HAS_ASYNC_LDS 0
#endif

__device__ __forceinline__ void rel_copy16_to_lds(const __bf16* g, __bf16* l)
{
#if REL_HAS_ASYNC_LDS
    __builtin_amdgcn_global_load_async_to_lds_b128(
        (__attribute__((address_space(1))) v4i*)g,
        (__attribute__((address_space(3))) v4i*)l,
        0, 0);
#else
    *(float4*)l = *(const float4*)g;
#endif
}

__device__ __forceinline__ void rel_wait_async_le4()
{
#if REL_HAS_ASYNC_LDS
#if __has_builtin(__builtin_amdgcn_s_wait_asynccnt)
    __builtin_amdgcn_s_wait_asynccnt(4);
#else
    asm volatile("s_wait_asynccnt 0x4" ::: "memory");
#endif
#endif
}

__device__ __forceinline__ void rel_wait_async_le0()
{
#if REL_HAS_ASYNC_LDS
#if __has_builtin(__builtin_amdgcn_s_wait_asynccnt)
    __builtin_amdgcn_s_wait_asynccnt(0);
#else
    asm volatile("s_wait_asynccnt 0x0" ::: "memory");
#endif
#endif
}

// ---------------------------------------------------------------------------
// Zero-fill (deterministic accumulator init each call; graph-capture safe)
// ---------------------------------------------------------------------------
__global__ void rel_zero_f4(float4* __restrict__ p, int n4)
{
    int i = blockIdx.x * blockDim.x + threadIdx.x;
    if (i < n4) p[i] = make_float4(0.f, 0.f, 0.f, 0.f);
}

// ---------------------------------------------------------------------------
// Stage 0: fused concat(sbj,obj) -> relu -> bf16, row-major [B][4096]
// ---------------------------------------------------------------------------
__global__ void rel_pack_x_relu_bf16(const float* __restrict__ sbj,
                                     const float* __restrict__ obj,
                                     __bf16* __restrict__ X, int n)
{
    int idx = blockIdx.x * blockDim.x + threadIdx.x;
    if (idx >= n) return;
    int b = idx >> 12;          // / 4096
    int k = idx & 4095;
    float v = (k < D_HALF) ? sbj[(size_t)b * D_HALF + k]
                           : obj[(size_t)b * D_HALF + (k - D_HALF)];
    X[idx] = (__bf16)fmaxf(v, 0.0f);
}

// ---------------------------------------------------------------------------
// Stage 0b: W [K][N] fp32 row-major  ->  Wt [N][K] bf16 (column-major of W)
// ---------------------------------------------------------------------------
__global__ void rel_transpose_to_bf16(const float* __restrict__ W,
                                      __bf16* __restrict__ Wt,
                                      int K, int N)
{
    __shared__ float tile[32][33];
    const int kT = blockIdx.x * 32;
    const int nT = blockIdx.y * 32;
    const int tx = threadIdx.x;     // 0..31
    const int ty = threadIdx.y;     // 0..7
#pragma unroll
    for (int i = 0; i < 32; i += 8)
        tile[ty + i][tx] = W[(size_t)(kT + ty + i) * N + nT + tx];
    __syncthreads();
#pragma unroll
    for (int i = 0; i < 32; i += 8)
        Wt[(size_t)(nT + ty + i) * K + kT + tx] = (__bf16)tile[tx][ty + i];
}

// ---------------------------------------------------------------------------
// LDS fragment loaders (ISA 7.12.2, wave32, 16-bit element layouts).
// Tiles in LDS mirror global layout, rows padded to LDS_LDK=40 elements
// (80B stride => per-lane b128 starts tile all 64 banks; conflict-free).
// A 16x32 (MxK): lane = row&15; half = lane>>4
//   v[0..7]  = As[row][half*8 .. +8] ; v[8..15] = As[row][16+half*8 .. +8]
// B 32x16 (KxN) from col-major tile Bs[n][k]: column n=lane&15, half = lane>>4
//   v[0..7]  = Bs[n][half*8 .. +8] ; v[8..15] = Bs[n][16+half*8 .. +8]
// ---------------------------------------------------------------------------
__device__ __forceinline__ v16bf rel_lds_a_frag(const __bf16* As, int mOff, int lane)
{
    const int r = lane & 15, h = lane >> 4;
    const __bf16* p = As + (mOff + r) * LDS_LDK + h * 8;
    union { float4 f; __bf16 e[8]; } u0, u1;
    u0.f = *(const float4*)(p);
    u1.f = *(const float4*)(p + 16);
    v16bf out;
#pragma unroll
    for (int j = 0; j < 8; ++j) { out[j] = u0.e[j]; out[j + 8] = u1.e[j]; }
    return out;
}

__device__ __forceinline__ v16bf rel_lds_b_frag(const __bf16* Bs, int nOff, int lane)
{
    const int c = lane & 15, h = lane >> 4;
    const __bf16* p = Bs + (nOff + c) * LDS_LDK + h * 8;
    union { float4 f[2]; __bf16 e[16]; } u;
    u.f[0] = *(const float4*)(p);
    u.f[1] = *(const float4*)(p + 16);
    v16bf out;
#pragma unroll
    for (int j = 0; j < 16; ++j) out[j] = u.e[j];
    return out;
}

// ---------------------------------------------------------------------------
// Stage a 128x32 A-tile and 128(n)x32(k) B-tile into LDS.
// 256 threads, each issues 2+2 async b128 copies (4 per wave -> ASYNCcnt+=4).
// ---------------------------------------------------------------------------
__device__ __forceinline__ void rel_stage_tile(const __bf16* __restrict__ A,
                                               const __bf16* __restrict__ Bt,
                                               int K, int mBase0, int nBase0, int kb,
                                               __bf16* As, __bf16* Bs, int tid)
{
#pragma unroll
    for (int s = 0; s < 2; ++s) {
        const int c   = s * 256 + tid;      // 16B chunk id, 0..511
        const int row = c >> 2;             // 0..127
        const int ko  = (c & 3) * 8;        // element offset within 32-wide row
        rel_copy16_to_lds(A + (size_t)(mBase0 + row) * K + kb + ko,
                          As + row * LDS_LDK + ko);
    }
#pragma unroll
    for (int s = 0; s < 2; ++s) {
        const int c   = s * 256 + tid;
        const int row = c >> 2;
        const int ko  = (c & 3) * 8;
        rel_copy16_to_lds(Bt + (size_t)(nBase0 + row) * K + kb + ko,
                          Bs + row * LDS_LDK + ko);
    }
}

// ---------------------------------------------------------------------------
// Split-K WMMA GEMM partial:  outAcc[M][N] += A_bf[M][kslice] @ Bt_bf^T
// Block = 256 threads = 8 waves; wave tile 64x32 (4x2 acc tiles of v8f);
// block tile 128x128; K step 32; gridDim.z = split-K factor.
// Double-buffered async global->LDS staging (ASYNCcnt), WMMA fed from LDS.
// Partials combined with GLOBAL_ATOMIC_ADD_F32 (non-returning, L2-rate).
// Full tiles, no divergence -> EXEC all-ones at every WMMA (ISA 7.12 req).
// LDS: 2 * (2 * 128*40*2B) = 40KB per block; 2 blocks/WGP = 80KB of 320KB.
// ---------------------------------------------------------------------------
__global__ __launch_bounds__(256, 2)
void rel_gemm_wmma_bf16_splitk(const __bf16* __restrict__ A,
                               const __bf16* __restrict__ Bt,
                               float* __restrict__ outAcc,
                               int M, int N, int K, int Kc)
{
    __shared__ __bf16 As[2][128 * LDS_LDK];
    __shared__ __bf16 Bs[2][128 * LDS_LDK];

    const int tid   = threadIdx.x;
    const int lane  = tid & 31;
    const int wid   = tid >> 5;
    const int waveM = wid & 1;          // 2 waves along M
    const int waveN = wid >> 1;         // 4 waves along N
    const int mBase0 = blockIdx.x * 128;
    const int nBase0 = blockIdx.y * 128;
    const int mW = waveM * 64;          // wave offsets inside LDS tile
    const int nW = waveN * 32;
    const int kLo = blockIdx.z * Kc;
    const int nt  = Kc / TILE_K;

    v8f acc[4][2];
#pragma unroll
    for (int i = 0; i < 4; ++i)
#pragma unroll
        for (int j = 0; j < 2; ++j)
            acc[i][j] = (v8f){0.f, 0.f, 0.f, 0.f, 0.f, 0.f, 0.f, 0.f};

    // Prologue: stage first tile
    rel_stage_tile(A, Bt, K, mBase0, nBase0, kLo, As[0], Bs[0], tid);

    for (int it = 0; it < nt; ++it) {
        const int cur = it & 1;
        if (it + 1 < nt) {
            rel_stage_tile(A, Bt, K, mBase0, nBase0, kLo + (it + 1) * TILE_K,
                           As[cur ^ 1], Bs[cur ^ 1], tid);
            rel_wait_async_le4();       // group `it` (first 4 in-order) landed
        } else {
            rel_wait_async_le0();
        }
        __syncthreads();                // all waves' stores visible

        v16bf a[4], b[2];
#pragma unroll
        for (int i = 0; i < 4; ++i)
            a[i] = rel_lds_a_frag(As[cur], mW + 16 * i, lane);
#pragma unroll
        for (int j = 0; j < 2; ++j)
            b[j] = rel_lds_b_frag(Bs[cur], nW + 16 * j, lane);
#pragma unroll
        for (int i = 0; i < 4; ++i)
#pragma unroll
            for (int j = 0; j < 2; ++j)
                acc[i][j] = __builtin_amdgcn_wmma_f32_16x16x32_bf16(
                    false, a[i], false, b[j], (short)0, acc[i][j], false, false);

        __syncthreads();                // compute done before buffer reuse
    }

    // Epilogue: straight-line atomic accumulation.
    // C/D layout: lane -> n = lane&15 ; VGPR r -> m = (lane>>4)*8 + r
    const int cN = lane & 15, cH = lane >> 4;
#pragma unroll
    for (int i = 0; i < 4; ++i) {
#pragma unroll
        for (int j = 0; j < 2; ++j) {
            const int n = nBase0 + nW + 16 * j + cN;
#pragma unroll
            for (int r = 0; r < 8; ++r) {
                const int m = mBase0 + mW + 16 * i + cH * 8 + r;
                atomicAdd(&outAcc[(size_t)m * N + n], acc[i][j][r]);
            }
        }
    }
}

// ---------------------------------------------------------------------------
// Finalize GEMM1: Hbf = bf16(relu(hidAcc + b1)), row-major [B][D_OUT]
// ---------------------------------------------------------------------------
__global__ void rel_finalize_hidden(const float* __restrict__ hidAcc,
                                    const float* __restrict__ b1,
                                    __bf16* __restrict__ Hbf, int n)
{
    int idx = blockIdx.x * blockDim.x + threadIdx.x;
    if (idx >= n) return;
    float v = hidAcc[idx] + b1[idx & (D_OUT - 1)];
    Hbf[idx] = (__bf16)fmaxf(v, 0.0f);
}

// ---------------------------------------------------------------------------
// Stage 3: scores[b][l] = -|| relu(label[l] - (embAcc[b] + b2)) ||_2
// ---------------------------------------------------------------------------
__global__ __launch_bounds__(N_LABELS)
void rel_order_sim_scores(const float* __restrict__ embAcc,
                          const float* __restrict__ b2,
                          const float* __restrict__ labels,
                          float* __restrict__ scores)
{
    __shared__ float se[D_OUT];
    const int b = blockIdx.x;
    const int t = threadIdx.x;            // label index, 0..127

    const float4* erow = (const float4*)(embAcc + (size_t)b * D_OUT);
    const float4* brow = (const float4*)b2;
    float4* s4 = (float4*)se;
#pragma unroll
    for (int i = t; i < D_OUT / 4; i += N_LABELS) {
        float4 e = erow[i], bb = brow[i];
        s4[i] = make_float4(e.x + bb.x, e.y + bb.y, e.z + bb.z, e.w + bb.w);
    }
    __syncthreads();

    const float* lrow = labels + (size_t)t * D_OUT;
    float a0 = 0.f, a1 = 0.f, a2 = 0.f, a3 = 0.f;
    for (int d = 0; d < D_OUT; d += 4) {
        const float4 lv = *(const float4*)(lrow + d);
        const float4 ev = *(const float4*)(se + d);
        float d0 = fmaxf(lv.x - ev.x, 0.f); a0 = fmaf(d0, d0, a0);
        float d1 = fmaxf(lv.y - ev.y, 0.f); a1 = fmaf(d1, d1, a1);
        float d2 = fmaxf(lv.z - ev.z, 0.f); a2 = fmaf(d2, d2, a2);
        float d3 = fmaxf(lv.w - ev.w, 0.f); a3 = fmaf(d3, d3, a3);
    }
    scores[(size_t)b * N_LABELS + t] = -sqrtf((a0 + a1) + (a2 + a3));
}

// ---------------------------------------------------------------------------
// Launch.  Workspace layout (28 MB):
//   [0,   8MB)  Xbf    : relu(concat) bf16      [1024][4096]
//   [8,  16MB)  W1t    : bf16 col-major         [1024][4096]
//   [16, 18MB)  Hbf    : relu(hidden) bf16      [1024][1024]
//   [18, 20MB)  W2t    : bf16 col-major         [1024][1024]
//   [20, 24MB)  hidAcc : fp32 split-K partials  [1024][1024]
//   [24, 28MB)  embAcc : fp32 split-K partials  [1024][1024]
// ---------------------------------------------------------------------------
extern "C" void kernel_launch(void* const* d_in, const int* in_sizes, int n_in,
                              void* d_out, int out_size, void* d_ws, size_t ws_size,
                              hipStream_t stream)
{
    const float* sbj    = (const float*)d_in[0];
    const float* obj    = (const float*)d_in[1];
    const float* W1     = (const float*)d_in[2];
    const float* b1     = (const float*)d_in[3];
    const float* W2     = (const float*)d_in[4];
    const float* b2     = (const float*)d_in[5];
    const float* labels = (const float*)d_in[6];
    float* scores       = (float*)d_out;

    char* ws = (char*)d_ws;
    __bf16* Xbf    = (__bf16*)(ws);
    __bf16* W1t    = (__bf16*)(ws + (8u  << 20));
    __bf16* Hbf    = (__bf16*)(ws + (16u << 20));
    __bf16* W2t    = (__bf16*)(ws + (18u << 20));
    float*  hidAcc = (float*) (ws + (20u << 20));
    float*  embAcc = (float*) (ws + (24u << 20));

    // Deterministic zero of both fp32 accumulators (2M floats = 512K float4)
    rel_zero_f4<<<(2 * B_SZ * D_OUT / 4 + 255) / 256, 256, 0, stream>>>(
        (float4*)hidAcc, 2 * B_SZ * D_OUT / 4);

    // Stage 0: activations + weight transposes (bf16 conversion)
    rel_pack_x_relu_bf16<<<(B_SZ * D_IN) / 256, 256, 0, stream>>>(sbj, obj, Xbf, B_SZ * D_IN);
    rel_transpose_to_bf16<<<dim3(D_IN  / 32, D_OUT / 32), dim3(32, 8), 0, stream>>>(W1, W1t, D_IN,  D_OUT);
    rel_transpose_to_bf16<<<dim3(D_OUT / 32, D_OUT / 32), dim3(32, 8), 0, stream>>>(W2, W2t, D_OUT, D_OUT);

    // Stage 1: hidAcc += X @ W1   (split-K = 4  -> 256 workgroups)
    rel_gemm_wmma_bf16_splitk<<<dim3(B_SZ / 128, D_OUT / 128, 4), 256, 0, stream>>>(
        Xbf, W1t, hidAcc, B_SZ, D_OUT, D_IN, D_IN / 4);
    // Hbf = bf16(relu(hidAcc + b1))
    rel_finalize_hidden<<<(B_SZ * D_OUT) / 256, 256, 0, stream>>>(
        hidAcc, b1, Hbf, B_SZ * D_OUT);

    // Stage 2: embAcc += H @ W2   (split-K = 2  -> 128 workgroups)
    rel_gemm_wmma_bf16_splitk<<<dim3(B_SZ / 128, D_OUT / 128, 2), 256, 0, stream>>>(
        Hbf, W2t, embAcc, B_SZ, D_OUT, D_OUT, D_OUT / 2);

    // Stage 3: order-sim scores (b2 folded into the LDS emb row)
    rel_order_sim_scores<<<B_SZ, N_LABELS, 0, stream>>>(embAcc, b2, labels, scores);
}
